// IGIB_27350351741541
// MI455X (gfx1250) — matrix operand order repaired and checked by reading, not compiled, not run on hardware
//
#include <hip/hip_runtime.h>
#include <hip/hip_bf16.h>
#include <stdint.h>

// ---------------------------------------------------------------------------
// Problem constants (match reference)
// ---------------------------------------------------------------------------
#define NU_N     8192
#define NV_N     8192
#define DD       300      // D
#define BB       256      // num graphs
#define EM_IT    3
#define TAU      1.0f     // contrastive temperature
#define H2       600      // 2*D  (Fu0/Fv0 width)
#define KCMP     1200     // 4*D  (compress input width, also 2H for set2set)
#define KPCMP    1216     // KCMP padded to mult of 32
#define KPH2     608      // H2 padded to mult of 32 (for h @ Whh^T)
#define NPD      304      // D padded to mult of 16
#define GATE     2400     // 4*2*D LSTM gate width
#define MAXSEG   96       // per-graph segment cap (E[n]=32, max ~55 @ B=256)
#define EPS_KL   1e-7f

typedef __attribute__((ext_vector_type(16))) __bf16 v16bf;
typedef __attribute__((ext_vector_type(8)))  float  v8f;

// ---------------------------------------------------------------------------
// Device helpers
// ---------------------------------------------------------------------------
__device__ inline float wave_sum(float v) {
#pragma unroll
  for (int off = 16; off > 0; off >>= 1) v += __shfl_xor(v, off, 32);
  return v;
}

__device__ inline float block_sum(float v, float* sh) {
  int t = threadIdx.x;
  sh[t] = v; __syncthreads();
  for (int off = blockDim.x >> 1; off > 0; off >>= 1) {
    if (t < off) sh[t] += sh[t + off];
    __syncthreads();
  }
  float r = sh[0]; __syncthreads();
  return r;
}

__device__ inline float block_max(float v, float* sh) {
  int t = threadIdx.x;
  sh[t] = v; __syncthreads();
  for (int off = blockDim.x >> 1; off > 0; off >>= 1) {
    if (t < off) sh[t] = fmaxf(sh[t], sh[t + off]);
    __syncthreads();
  }
  float r = sh[0]; __syncthreads();
  return r;
}

__device__ inline float sigmoidf(float x) { return 1.0f / (1.0f + expf(-x)); }

// Stateless counter hash -> uniform in [2^-25, 1-2^-25]
__device__ inline float rng_u(uint32_t a, uint32_t b, uint32_t c) {
  uint32_t h = a * 0x9E3779B9u ^ (b + 0x85EBCA6Bu) * 0xC2B2AE35u ^ (c * 0x27D4EB2Fu);
  h ^= h >> 16; h *= 0x7FEB352Du; h ^= h >> 15; h *= 0x846CA68Bu; h ^= h >> 16;
  return ((h >> 8) + 0.5f) * (1.0f / 16777216.0f);
}

// ---------------------------------------------------------------------------
// Utility kernels
// ---------------------------------------------------------------------------
__global__ void k_zero_f32(float* p, int n) {
  for (int i = blockIdx.x * blockDim.x + threadIdx.x; i < n;
       i += gridDim.x * blockDim.x) p[i] = 0.0f;
}
__global__ void k_zero_i32(int* p, int n) {
  for (int i = blockIdx.x * blockDim.x + threadIdx.x; i < n;
       i += gridDim.x * blockDim.x) p[i] = 0;
}

// Row L2-normalize (one wave32 per row)
__global__ void k_rownorm(const float* __restrict__ x, float* __restrict__ y,
                          int n, int d) {
  int wave = (blockIdx.x * blockDim.x + threadIdx.x) >> 5;
  int lane = threadIdx.x & 31;
  if (wave >= n) return;
  const float* xr = x + (size_t)wave * d;
  float s = 0.f;
  for (int j = lane; j < d; j += 32) { float v = xr[j]; s += v * v; }
  s = wave_sum(s);
  float inv = 1.0f / fmaxf(sqrtf(s), 1e-12f);
  float* yr = y + (size_t)wave * d;
  for (int j = lane; j < d; j += 32) yr[j] = xr[j] * inv;
}

// Segment [start,end) from a sorted batch-index array
__global__ void k_seg_bounds(const int* __restrict__ batch, int n,
                             int* __restrict__ st, int* __restrict__ en) {
  int i = blockIdx.x * blockDim.x + threadIdx.x;
  if (i >= n) return;
  int b = batch[i];
  if (i == 0 || batch[i - 1] != b) st[b] = i;
  if (i == n - 1 || batch[i + 1] != b) en[b] = i + 1;
}

// ---------------------------------------------------------------------------
// Block-diagonal interaction: per graph b,
//   S = Hu_b Hv_b^T (kept in LDS),  Fu0 = [Hu | S Hv_b],  Fv0 = [Hv | S^T Hu_b]
// FLOPs here are tiny after exploiting the mask's block structure; VALU is fine.
// ---------------------------------------------------------------------------
__global__ void k_interact(const float* __restrict__ Hu, const float* __restrict__ Hv,
                           const int* __restrict__ us_, const int* __restrict__ ue_,
                           const int* __restrict__ vs_, const int* __restrict__ ve_,
                           float* __restrict__ Fu, float* __restrict__ Fv) {
  __shared__ float S[MAXSEG * MAXSEG];   // 36 KB of the WGP's 320 KB LDS
  int b = blockIdx.x;
  int us = us_[b], ue = ue_[b], vs = vs_[b], ve = ve_[b];
  int nuf = ue - us, nvf = ve - vs;
  int nu = min(nuf, MAXSEG), nv = min(nvf, MAXSEG);
  int tid = threadIdx.x;

  for (int p = tid; p < nu * nv; p += blockDim.x) {
    int u = p / nv, v = p % nv;
    const float* a = Hu + (size_t)(us + u) * DD;
    const float* c = Hv + (size_t)(vs + v) * DD;
    float s = 0.f;
    for (int k = 0; k < DD; ++k) s += a[k] * c[k];
    S[u * nv + v] = s;
  }
  __syncthreads();

  for (int p = tid; p < nuf * DD; p += blockDim.x) {
    int u = p / DD, d = p % DD;
    float acc = 0.f;
    if (u < nu)
      for (int v = 0; v < nv; ++v)
        acc += S[u * nv + v] * Hv[(size_t)(vs + v) * DD + d];
    float* fr = Fu + (size_t)(us + u) * H2;
    fr[d] = Hu[(size_t)(us + u) * DD + d];
    fr[DD + d] = acc;
  }
  for (int p = tid; p < nvf * DD; p += blockDim.x) {
    int v = p / DD, d = p % DD;
    float acc = 0.f;
    if (v < nv)
      for (int u = 0; u < nu; ++u)
        acc += S[u * nv + v] * Hu[(size_t)(us + u) * DD + d];
    float* fr = Fv + (size_t)(vs + v) * H2;
    fr[d] = Hv[(size_t)(vs + v) * DD + d];
    fr[DD + d] = acc;
  }
}

// Per-graph mean (and optional unbiased std) over dim columns; one block/graph
__global__ void k_seg_stats(const float* __restrict__ F,
                            const int* __restrict__ st, const int* __restrict__ en,
                            float* __restrict__ mean, float* __restrict__ stdo,
                            int dim) {
  int b = blockIdx.x;
  int s = st[b], e = en[b];
  float cnt = (float)(e - s);
  for (int d = threadIdx.x; d < dim; d += blockDim.x) {
    float sum = 0.f, sq = 0.f;
    for (int r = s; r < e; ++r) {
      float v = F[(size_t)r * dim + d];
      sum += v; sq += v * v;
    }
    float m = sum / fmaxf(cnt, 1.0f);
    mean[(size_t)b * dim + d] = m;
    if (stdo) {
      float var = (sq - cnt * m * m) / fmaxf(cnt - 1.0f, 1.0f);
      stdo[(size_t)b * dim + d] = sqrtf(fmaxf(var, 0.f));
    }
  }
}

// ---------------------------------------------------------------------------
// bf16 packing (zero-padded) for WMMA operands
// ---------------------------------------------------------------------------
__global__ void k_pack_bf16(const float* __restrict__ in, __bf16* __restrict__ out,
                            int M, int K, int Mp, int Kp) {
  int total = Mp * Kp;
  for (int i = blockIdx.x * blockDim.x + threadIdx.x; i < total;
       i += gridDim.x * blockDim.x) {
    int r = i / Kp, c = i % Kp;
    float v = (r < M && c < K) ? in[(size_t)r * K + c] : 0.0f;
    out[i] = (__bf16)v;
  }
}

// Compress input: X = [F(600) | gv[batch](600)] -> bf16 [n x KPCMP]
__global__ void k_build_cmp_in(const float* __restrict__ F, const float* __restrict__ gv,
                               const int* __restrict__ batch, __bf16* __restrict__ out,
                               int n) {
  int total = n * KPCMP;
  for (int i = blockIdx.x * blockDim.x + threadIdx.x; i < total;
       i += gridDim.x * blockDim.x) {
    int r = i / KPCMP, c = i % KPCMP;
    float v = 0.f;
    if (c < H2)        v = F[(size_t)r * H2 + c];
    else if (c < KCMP) v = gv[(size_t)batch[r] * H2 + (c - H2)];
    out[i] = (__bf16)v;
  }
}

// ---------------------------------------------------------------------------
// Generic WMMA GEMM:  C[M x N] = A[M x Kp] * Bt[N x Kp]^T   (bf16 in, f32 acc)
// One wave per 16x16 output tile; fragment lane layout per ISA 7.12.2:
//   A 16-bit 16x32: lanes 0-15 row M=lane hold K {0..7,16..23}; lanes 16-31 +8
//   B 16-bit 32x16: lanes 0-15 col N=lane hold K {0..7,16..23}; lanes 16-31 +8
//   C/D f32: lane<16 -> M=r, lane>=16 -> M=8+r; N = lane&15
// A must be padded to tilesM*16 rows, Bt to tilesN*16 rows (packers zero-pad).
// ---------------------------------------------------------------------------
__global__ void k_wmma_gemm_bf16(const __bf16* __restrict__ A,
                                 const __bf16* __restrict__ Bt,
                                 float* __restrict__ C,
                                 int M, int N, int Kp, int ldc,
                                 int tilesN, int nTiles) {
  int wave = (blockIdx.x * blockDim.x + threadIdx.x) >> 5;
  if (wave >= nTiles) return;
  int lane = threadIdx.x & 31;
  int mt = (wave / tilesN) * 16;
  int nt = (wave % tilesN) * 16;

  const __bf16* arow = A  + (size_t)(mt + (lane & 15)) * Kp + ((lane < 16) ? 0 : 8);
  const __bf16* brow = Bt + (size_t)(nt + (lane & 15)) * Kp + ((lane < 16) ? 0 : 8);

  v8f acc = {};
  for (int k0 = 0; k0 < Kp; k0 += 32) {
    __builtin_prefetch(arow + k0 + 128, 0, 1);   // global_prefetch_b8
    __builtin_prefetch(brow + k0 + 128, 0, 1);
    v16bf a, b;
#pragma unroll
    for (int e = 0; e < 8; ++e) {
      a[e]     = arow[k0 + e];
      a[8 + e] = arow[k0 + 16 + e];
    }
#pragma unroll
    for (int e = 0; e < 8; ++e) { b[e] = brow[k0 + e]; b[8 + e] = brow[k0 + 16 + e]; }
    acc = __builtin_amdgcn_wmma_f32_16x16x32_bf16(
        false, a, false, b, (short)0, acc, false, false);
  }

  int cn = nt + (lane & 15);
#pragma unroll
  for (int r = 0; r < 8; ++r) {
    int cm = mt + ((lane < 16) ? r : r + 8);
    if (cm < M && cn < N) C[(size_t)cm * ldc + cn] = acc[r];
  }
}

// ---------------------------------------------------------------------------
// Compress tail: BatchNorm (batch stats) -> ReLU -> linear(300->1) -> noisy sigmoid
// ---------------------------------------------------------------------------
__global__ void k_bn_stats(const float* __restrict__ H, float* __restrict__ mu,
                           float* __restrict__ var, int n, int ldc) {
  __shared__ float sh[256];
  int c = blockIdx.x;
  float s = 0.f, q = 0.f;
  for (int r = threadIdx.x; r < n; r += blockDim.x) {
    float v = H[(size_t)r * ldc + c];
    s += v; q += v * v;
  }
  float ts = block_sum(s, sh);
  float tq = block_sum(q, sh);
  if (threadIdx.x == 0) {
    float m = ts / (float)n;
    mu[c] = m;
    var[c] = fmaxf(tq / (float)n - m * m, 0.f);   // biased, as jnp.var
  }
}

// Note: cb1 is omitted on purpose — a per-column constant bias cancels exactly
// in BatchNorm's (h - mean(h)) since batch stats are computed on the same h.
__global__ void k_cmp_finish(const float* __restrict__ Hl, const float* __restrict__ mu,
                             const float* __restrict__ var, const float* __restrict__ g,
                             const float* __restrict__ be, const float* __restrict__ W2,
                             const float* __restrict__ b2, float* __restrict__ lam,
                             int n, int ldc, uint32_t salt) {
  int wave = (blockIdx.x * blockDim.x + threadIdx.x) >> 5;
  int lane = threadIdx.x & 31;
  if (wave >= n) return;
  const float* hr = Hl + (size_t)wave * ldc;
  float p = 0.f;
  for (int d = lane; d < DD; d += 32) {
    float h = (hr[d] - mu[d]) * rsqrtf(var[d] + 1e-5f) * g[d] + be[d];
    p += fmaxf(h, 0.f) * W2[d];
  }
  p = wave_sum(p) + b2[0];
  if (lane == 0) {
    const float bc = 1e-4f;
    float u = rng_u((uint32_t)wave, salt, 0xBEEFu);
    float eps = (2.f * bc - 1.f) * u + (1.f - bc);     // uniform in (bc, 1-bc)
    float z = logf(eps) - logf(1.f - eps) + p;
    lam[wave] = sigmoidf(z);
  }
}

// noisy = lam*F + (1-lam)*mean  +  U * (1-lam)*std
__global__ void k_noisy(const float* __restrict__ F, const float* __restrict__ mean,
                        const float* __restrict__ stdv, const float* __restrict__ lam,
                        const int* __restrict__ batch, float* __restrict__ noisy,
                        int n, uint32_t salt) {
  int total = n * H2;
  for (int i = blockIdx.x * blockDim.x + threadIdx.x; i < total;
       i += gridDim.x * blockDim.x) {
    int r = i / H2, d = i % H2;
    int b = batch[r];
    float l = lam[r];
    float m = mean[(size_t)b * H2 + d];
    float s = stdv[(size_t)b * H2 + d];
    float nm = l * F[i] + (1.f - l) * m;
    float ns = (1.f - l) * s;
    noisy[i] = nm + rng_u((uint32_t)r, (uint32_t)d, salt) * ns;
  }
}

// ---------------------------------------------------------------------------
// Set2Set pieces (LSTM gates come from the WMMA GEMM; order i,f,g,o)
// ---------------------------------------------------------------------------
__global__ void k_lstm(const float* __restrict__ G1, const float* __restrict__ G2,
                       const float* __restrict__ bih, const float* __restrict__ bhh,
                       float* __restrict__ h, float* __restrict__ c, int nb) {
  int total = nb * H2;
  for (int i = blockIdx.x * blockDim.x + threadIdx.x; i < total;
       i += gridDim.x * blockDim.x) {
    int b = i / H2, d = i % H2;
    size_t base = (size_t)b * GATE;
    float gi = G1[base + d]            + G2[base + d]            + bih[d]            + bhh[d];
    float gf = G1[base + H2 + d]       + G2[base + H2 + d]       + bih[H2 + d]       + bhh[H2 + d];
    float gg = G1[base + 2 * H2 + d]   + G2[base + 2 * H2 + d]   + bih[2 * H2 + d]   + bhh[2 * H2 + d];
    float go = G1[base + 3 * H2 + d]   + G2[base + 3 * H2 + d]   + bih[3 * H2 + d]   + bhh[3 * H2 + d];
    float cc = sigmoidf(gf) * c[i] + sigmoidf(gi) * tanhf(gg);
    c[i] = cc;
    h[i] = sigmoidf(go) * tanhf(cc);
  }
}

__global__ void k_attn_e(const float* __restrict__ x, const float* __restrict__ h,
                         const int* __restrict__ batch, float* __restrict__ e, int n) {
  int wave = (blockIdx.x * blockDim.x + threadIdx.x) >> 5;
  int lane = threadIdx.x & 31;
  if (wave >= n) return;
  const float* xr = x + (size_t)wave * H2;
  const float* hr = h + (size_t)batch[wave] * H2;
  float s = 0.f;
  for (int d = lane; d < H2; d += 32) s += xr[d] * hr[d];
  s = wave_sum(s);
  if (lane == 0) e[wave] = s;
}

__global__ void k_seg_softmax(const float* __restrict__ e, const int* __restrict__ st,
                              const int* __restrict__ en, float* __restrict__ a) {
  __shared__ float sh[256];
  int b = blockIdx.x;
  int s = st[b], t = en[b];
  float mx = -INFINITY;
  for (int r = s + threadIdx.x; r < t; r += blockDim.x) mx = fmaxf(mx, e[r]);
  float m = block_max(mx, sh);
  if (!(m > -1e37f)) m = 0.f;            // jnp: where(isfinite(max), max, 0)
  float sum = 0.f;
  for (int r = s + threadIdx.x; r < t; r += blockDim.x) sum += expf(e[r] - m);
  float tot = block_sum(sum, sh);
  float inv = 1.f / fmaxf(tot, 1e-16f);
  for (int r = s + threadIdx.x; r < t; r += blockDim.x) a[r] = expf(e[r] - m) * inv;
}

__global__ void k_wsegsum(const float* __restrict__ a, const float* __restrict__ x,
                          const int* __restrict__ st, const int* __restrict__ en,
                          float* __restrict__ rout) {
  int b = blockIdx.x;
  int s = st[b], e = en[b];
  for (int d = threadIdx.x; d < H2; d += blockDim.x) {
    float acc = 0.f;
    for (int r = s; r < e; ++r) acc += a[r] * x[(size_t)r * H2 + d];
    rout[(size_t)b * H2 + d] = acc;
  }
}

__global__ void k_build_qstar(const float* __restrict__ h, const float* __restrict__ r,
                              float* __restrict__ q, int nb) {
  int total = nb * KCMP;
  for (int i = blockIdx.x * blockDim.x + threadIdx.x; i < total;
       i += gridDim.x * blockDim.x) {
    int b = i / KCMP, d = i % KCMP;
    q[i] = (d < H2) ? h[(size_t)b * H2 + d] : r[(size_t)b * H2 + (d - H2)];
  }
}

// ---------------------------------------------------------------------------
// Heads: contrastive (both directions from one logits matrix), preds, KL
// ---------------------------------------------------------------------------
__global__ void k_contrastive(const float* __restrict__ logits,
                              float* __restrict__ c1, float* __restrict__ c2) {
  __shared__ float sh[256];
  int t = threadIdx.x;
  const float inv_tau = 1.0f / TAU;
  float diag = logits[t * BB + t] * inv_tau;

  float mx = -INFINITY;
  for (int j = 0; j < BB; ++j) mx = fmaxf(mx, logits[t * BB + j] * inv_tau);
  float se = 0.f;
  for (int j = 0; j < BB; ++j) se += expf(logits[t * BB + j] * inv_tau - mx);
  float row_term = (mx + logf(se)) - diag;

  float mc = -INFINITY;
  for (int j = 0; j < BB; ++j) mc = fmaxf(mc, logits[j * BB + t] * inv_tau);
  float sc = 0.f;
  for (int j = 0; j < BB; ++j) sc += expf(logits[j * BB + t] * inv_tau - mc);
  float col_term = (mc + logf(sc)) - diag;

  float s1 = block_sum(row_term, sh);
  float s2 = block_sum(col_term, sh);
  if (t == 0) { *c1 = s1 / (float)BB; *c2 = s2 / (float)BB; }
}

__global__ void k_preds(const float* __restrict__ su, const float* __restrict__ sv,
                        const float* __restrict__ pW, const float* __restrict__ pb,
                        float* __restrict__ out) {
  int wave = (blockIdx.x * blockDim.x + threadIdx.x) >> 5;
  int lane = threadIdx.x & 31;
  if (wave >= BB) return;
  float s = 0.f;
  for (int d = lane; d < KCMP; d += 32) s += su[(size_t)wave * KCMP + d] * pW[d];
  for (int d = lane; d < KCMP; d += 32) s += sv[(size_t)wave * KCMP + d] * pW[KCMP + d];
  s = wave_sum(s);
  if (lane == 0) out[wave] = s + pb[0];
}

// mean(kl) = (1/(B*600)) * sum_{n,d} [ 0.5*((1-lam)*std/(std+E))^2
//                                      + (lam*(F-mean)/(std+E))^2 ]
__global__ void k_kl(const float* __restrict__ F, const float* __restrict__ mean,
                     const float* __restrict__ stdv, const float* __restrict__ lam,
                     const int* __restrict__ batch, float* __restrict__ slot,
                     float scale, int n) {
  __shared__ float sh[256];
  int total = n * H2;
  float acc = 0.f;
  for (int i = blockIdx.x * blockDim.x + threadIdx.x; i < total;
       i += gridDim.x * blockDim.x) {
    int r = i / H2, d = i % H2;
    int b = batch[r];
    float l = lam[r];
    float m = mean[(size_t)b * H2 + d];
    float s = stdv[(size_t)b * H2 + d];
    float ratio = (1.f - l) * s / (s + EPS_KL);
    float z = l * (F[i] - m) / (s + EPS_KL);
    acc += 0.5f * ratio * ratio + z * z;
  }
  float tot = block_sum(acc, sh);
  if (threadIdx.x == 0) atomicAdd(slot, tot * scale);
}

// ---------------------------------------------------------------------------
// Host orchestration
// ---------------------------------------------------------------------------
static inline size_t alignup(size_t x) { return (x + 255) & ~(size_t)255; }

struct Bump {
  char* p;
  void* take(size_t bytes) { void* r = p; p += alignup(bytes); return r; }
};

static void launch_gemm(const __bf16* A, const __bf16* Bt, float* C,
                        int M, int N, int Kp, int ldc, hipStream_t stream) {
  int tilesM = (M + 15) / 16, tilesN = (N + 15) / 16;
  int nTiles = tilesM * tilesN;
  int blocks = (nTiles + 7) / 8;                 // 8 waves (256 threads) / block
  k_wmma_gemm_bf16<<<blocks, 256, 0, stream>>>(A, Bt, C, M, N, Kp, ldc, tilesN, nTiles);
}

extern "C" void kernel_launch(void* const* d_in, const int* in_sizes, int n_in,
                              void* d_out, int out_size, void* d_ws, size_t ws_size,
                              hipStream_t stream) {
  (void)in_sizes; (void)n_in; (void)out_size; (void)ws_size;
  const float* solute_x  = (const float*)d_in[0];
  const float* solvent_x = (const float*)d_in[1];
  const float* cW1   = (const float*)d_in[2];
  // d_in[3] = cb1 (cancels inside BatchNorm; see k_cmp_finish note)
  const float* cgam  = (const float*)d_in[4];
  const float* cbeta = (const float*)d_in[5];
  const float* cW2   = (const float*)d_in[6];
  const float* cb2   = (const float*)d_in[7];
  const float* uWih  = (const float*)d_in[8];
  const float* uWhh  = (const float*)d_in[9];
  const float* ubih  = (const float*)d_in[10];
  const float* ubhh  = (const float*)d_in[11];
  const float* vWih  = (const float*)d_in[12];
  const float* vWhh  = (const float*)d_in[13];
  const float* vbih  = (const float*)d_in[14];
  const float* vbhh  = (const float*)d_in[15];
  const float* pW    = (const float*)d_in[16];
  const float* pb    = (const float*)d_in[17];
  const int* ub = (const int*)d_in[18];
  const int* vb = (const int*)d_in[19];

  float* out = (float*)d_out;                      // [preds(256), kl1, kl2, c1, c2]

  Bump ws{(char*)d_ws};
  float* Hu    = (float*)ws.take(sizeof(float) * NU_N * DD);
  float* Hv    = (float*)ws.take(sizeof(float) * NV_N * DD);
  int*   segI  = (int*)ws.take(sizeof(int) * 4 * BB);
  int *us = segI, *ue = segI + BB, *vs = segI + 2 * BB, *ve = segI + 3 * BB;
  float* Fu    = (float*)ws.take(sizeof(float) * NU_N * H2);
  float* Fv    = (float*)ws.take(sizeof(float) * NV_N * H2);
  float* mean_u = (float*)ws.take(sizeof(float) * BB * H2);
  float* std_u  = (float*)ws.take(sizeof(float) * BB * H2);
  float* mean_v = (float*)ws.take(sizeof(float) * BB * H2);   // == solvent_gv
  float* std_v  = (float*)ws.take(sizeof(float) * BB * H2);
  float* gv_u   = (float*)ws.take(sizeof(float) * BB * H2);   // solute_gv per iter
  __bf16* Xbf  = (__bf16*)ws.take(sizeof(__bf16) * NU_N * KPCMP);
  __bf16* W1bf = (__bf16*)ws.take(sizeof(__bf16) * NPD * KPCMP);
  float* Hlin  = (float*)ws.take(sizeof(float) * NU_N * NPD);
  float* bn_mu = (float*)ws.take(sizeof(float) * NPD);
  float* bn_var= (float*)ws.take(sizeof(float) * NPD);
  float* lam_u = (float*)ws.take(sizeof(float) * NU_N);
  float* lam_v = (float*)ws.take(sizeof(float) * NV_N);
  float* noisy_u = (float*)ws.take(sizeof(float) * NU_N * H2);
  float* noisy_v = (float*)ws.take(sizeof(float) * NV_N * H2);
  // set2set scratch
  float* s2_h  = (float*)ws.take(sizeof(float) * BB * H2);
  float* s2_c  = (float*)ws.take(sizeof(float) * BB * H2);
  float* qs_u  = (float*)ws.take(sizeof(float) * BB * KCMP);
  float* qs_v  = (float*)ws.take(sizeof(float) * BB * KCMP);
  float* G1    = (float*)ws.take(sizeof(float) * BB * GATE);
  float* G2    = (float*)ws.take(sizeof(float) * BB * GATE);
  __bf16* qs_bf  = (__bf16*)ws.take(sizeof(__bf16) * BB * KPCMP);
  __bf16* h_bf   = (__bf16*)ws.take(sizeof(__bf16) * BB * KPH2);
  __bf16* Wih_bf = (__bf16*)ws.take(sizeof(__bf16) * GATE * KPCMP);
  __bf16* Whh_bf = (__bf16*)ws.take(sizeof(__bf16) * GATE * KPH2);
  float* ebuf  = (float*)ws.take(sizeof(float) * NU_N);
  float* abuf  = (float*)ws.take(sizeof(float) * NU_N);
  float* rbuf  = (float*)ws.take(sizeof(float) * BB * H2);
  float* sn_u  = (float*)ws.take(sizeof(float) * BB * KCMP);
  float* sn_v  = (float*)ws.take(sizeof(float) * BB * KCMP);
  __bf16* snu_bf = (__bf16*)ws.take(sizeof(__bf16) * BB * KPCMP);
  __bf16* snv_bf = (__bf16*)ws.take(sizeof(__bf16) * BB * KPCMP);
  float* logits = (float*)ws.take(sizeof(float) * BB * BB);

  // ---- out slots 256..259 need zero (kl accumulated via atomics) ----
  k_zero_f32<<<1, 32, 0, stream>>>(out + BB, 4);

  // ---- normalize inputs ----
  k_rownorm<<<(NU_N * 32) / 256, 256, 0, stream>>>(solute_x,  Hu, NU_N, DD);
  k_rownorm<<<(NV_N * 32) / 256, 256, 0, stream>>>(solvent_x, Hv, NV_N, DD);

  // ---- segment bounds ----
  k_zero_i32<<<4, 256, 0, stream>>>(segI, 4 * BB);
  k_seg_bounds<<<NU_N / 256, 256, 0, stream>>>(ub, NU_N, us, ue);
  k_seg_bounds<<<NV_N / 256, 256, 0, stream>>>(vb, NV_N, vs, ve);

  // ---- block-diagonal interaction -> Fu0, Fv0 ----
  k_interact<<<BB, 256, 0, stream>>>(Hu, Hv, us, ue, vs, ve, Fu, Fv);

  // ---- per-graph stats ----
  k_seg_stats<<<BB, 256, 0, stream>>>(Fu, us, ue, mean_u, std_u, H2);
  k_seg_stats<<<BB, 256, 0, stream>>>(Fv, vs, ve, mean_v, std_v, H2);  // mean_v == solvent_gv

  // ---- pack compress weight once ----
  k_pack_bf16<<<512, 256, 0, stream>>>(cW1, W1bf, DD, KCMP, NPD, KPCMP);

  // ---- EM loop ----
  for (int it = 0; it < EM_IT; ++it) {
    uint32_t salt = 0x5EED0000u + 16u * (uint32_t)it;
    // lam_u from [Fu0 | solvent_gv]
    k_build_cmp_in<<<2048, 256, 0, stream>>>(Fu, mean_v, ub, Xbf, NU_N);
    launch_gemm(Xbf, W1bf, Hlin, NU_N, DD, KPCMP, NPD, stream);
    k_bn_stats<<<DD, 256, 0, stream>>>(Hlin, bn_mu, bn_var, NU_N, NPD);
    k_cmp_finish<<<(NU_N * 32) / 256, 256, 0, stream>>>(
        Hlin, bn_mu, bn_var, cgam, cbeta, cW2, cb2, lam_u, NU_N, NPD, salt + 0);
    k_noisy<<<2048, 256, 0, stream>>>(Fu, mean_u, std_u, lam_u, ub, noisy_u, NU_N, salt + 1);
    k_seg_stats<<<BB, 256, 0, stream>>>(noisy_u, us, ue, gv_u, nullptr, H2);
    // lam_v from [Fv0 | solute_gv]
    k_build_cmp_in<<<2048, 256, 0, stream>>>(Fv, gv_u, vb, Xbf, NV_N);
    launch_gemm(Xbf, W1bf, Hlin, NV_N, DD, KPCMP, NPD, stream);
    k_bn_stats<<<DD, 256, 0, stream>>>(Hlin, bn_mu, bn_var, NV_N, NPD);
    k_cmp_finish<<<(NV_N * 32) / 256, 256, 0, stream>>>(
        Hlin, bn_mu, bn_var, cgam, cbeta, cW2, cb2, lam_v, NV_N, NPD, salt + 2);
    k_noisy<<<2048, 256, 0, stream>>>(Fv, mean_v, std_v, lam_v, vb, noisy_v, NV_N, salt + 3);
  }

  // ---- Set2Set (2 steps) for each side ----
  for (int side = 0; side < 2; ++side) {
    const float* x    = side ? noisy_v : noisy_u;
    const int*  batch = side ? vb : ub;
    const int*  st    = side ? vs : us;
    const int*  en    = side ? ve : ue;
    const float* Wih  = side ? vWih : uWih;
    const float* Whh  = side ? vWhh : uWhh;
    const float* bih  = side ? vbih : ubih;
    const float* bhh  = side ? vbhh : ubhh;
    float* qs = side ? qs_v : qs_u;
    int n = side ? NV_N : NU_N;

    k_pack_bf16<<<2048, 256, 0, stream>>>(Wih, Wih_bf, GATE, KCMP, GATE, KPCMP);
    k_pack_bf16<<<1024, 256, 0, stream>>>(Whh, Whh_bf, GATE, H2,   GATE, KPH2);
    k_zero_f32<<<256, 256, 0, stream>>>(s2_h, BB * H2);
    k_zero_f32<<<256, 256, 0, stream>>>(s2_c, BB * H2);
    k_zero_f32<<<256, 256, 0, stream>>>(qs, BB * KCMP);

    for (int step = 0; step < 2; ++step) {
      k_pack_bf16<<<512, 256, 0, stream>>>(qs,   qs_bf, BB, KCMP, BB, KPCMP);
      k_pack_bf16<<<256, 256, 0, stream>>>(s2_h, h_bf,  BB, H2,   BB, KPH2);
      launch_gemm(qs_bf, Wih_bf, G1, BB, GATE, KPCMP, GATE, stream);
      launch_gemm(h_bf,  Whh_bf, G2, BB, GATE, KPH2,  GATE, stream);
      k_lstm<<<512, 256, 0, stream>>>(G1, G2, bih, bhh, s2_h, s2_c, BB);
      k_attn_e<<<(n * 32) / 256, 256, 0, stream>>>(x, s2_h, batch, ebuf, n);
      k_seg_softmax<<<BB, 256, 0, stream>>>(ebuf, st, en, abuf);
      k_wsegsum<<<BB, 256, 0, stream>>>(abuf, x, st, en, rbuf);
      k_build_qstar<<<512, 256, 0, stream>>>(s2_h, rbuf, qs, BB);
    }
  }

  // ---- contrastive heads ----
  k_rownorm<<<32, 256, 0, stream>>>(qs_u, sn_u, BB, KCMP);
  k_rownorm<<<32, 256, 0, stream>>>(qs_v, sn_v, BB, KCMP);
  k_pack_bf16<<<512, 256, 0, stream>>>(sn_u, snu_bf, BB, KCMP, BB, KPCMP);
  k_pack_bf16<<<512, 256, 0, stream>>>(sn_v, snv_bf, BB, KCMP, BB, KPCMP);
  launch_gemm(snu_bf, snv_bf, logits, BB, BB, KPCMP, BB, stream);
  k_contrastive<<<1, 256, 0, stream>>>(logits, out + BB + 2, out + BB + 3);

  // ---- preds ----
  k_preds<<<32, 256, 0, stream>>>(qs_u, qs_v, pW, pb, out);

  // ---- KL terms (last-iteration lam; nmean/nstd recomputed analytically) ----
  float kscale = 1.0f / ((float)BB * (float)H2);
  k_kl<<<512, 256, 0, stream>>>(Fu, mean_u, std_u, lam_u, ub, out + BB + 0, kscale, NU_N);
  k_kl<<<512, 256, 0, stream>>>(Fv, mean_v, std_v, lam_v, vb, out + BB + 1, kscale, NV_N);
}
